// SelfAttentionAlt_44186623541814
// MI455X (gfx1250) — matrix operand order repaired and checked by reading, not compiled, not run on hardware
//
#include <hip/hip_runtime.h>
#include <hip/hip_bf16.h>

// ---------------------------------------------------------------------------
// Self-attention (EMB=1024, HEADS=16, S=64, b=2, t=2048) for gfx1250 (MI455X).
// All GEMM stages use v_wmma_f32_16x16x32_bf16 (wave32), 32x64 tiles/wave
// (fits in <256 VGPRs: 64 acc + 96 double-buffered fragments), register
// double-buffered K loop + global_prefetch. Flash-style attention.
// ---------------------------------------------------------------------------

typedef __bf16 bf16_t;
typedef __attribute__((ext_vector_type(16))) __bf16 v16bf;
typedef __attribute__((ext_vector_type(8)))  __bf16 v8bf;
typedef __attribute__((ext_vector_type(8)))  float   v8f;

#define EMB   1024
#define HEADS 16
#define SDIM  64
#define BATCH 2
#define SEQ   2048
#define NTOK  (BATCH * SEQ)   // 4096

// f32 -> bf16, round to nearest even (no dependence on __bf16 arithmetic)
__device__ __forceinline__ bf16_t f2bf(float f) {
    union { float f; unsigned u; } c; c.f = f;
    unsigned r = (c.u + 0x7fffu + ((c.u >> 16) & 1u)) >> 16;
    unsigned short s = (unsigned short)r;
    return __builtin_bit_cast(bf16_t, s);
}

// A-fragment (16x32 bf16, MxK). Lane L<16: row m0+L, K = {k0+half*8..+7, k0+16+half*8..+7}
__device__ __forceinline__ v16bf load_frag_a(const bf16_t* base, int m0, int ld,
                                             int k0, int lane) {
    int half = lane >> 4;
    int row  = m0 + (lane & 15);
    const bf16_t* p = base + (size_t)row * ld + k0 + half * 8;
    v8bf lo = *(const v8bf*)(p);
    v8bf hi = *(const v8bf*)(p + 16);
    return __builtin_shufflevector(lo, hi, 0,1,2,3,4,5,6,7,8,9,10,11,12,13,14,15);
}

// B-fragment (32x16 bf16, KxN) from B^T storage [N][K]. Lane L: col n0+(L&15),
// K = {k0+half*8..+7, k0+16+half*8..+7}.
__device__ __forceinline__ v16bf load_frag_b(const bf16_t* baseT, int n0, int ld,
                                             int k0, int lane) {
    int half = lane >> 4;
    int col  = n0 + (lane & 15);
    const bf16_t* p = baseT + (size_t)col * ld + k0 + half * 8;
    v8bf lo = *(const v8bf*)(p);
    v8bf hi = *(const v8bf*)(p + 16);
    return __builtin_shufflevector(lo, hi, 0,1,2,3,4,5,6,7,8,9,10,11,12,13,14,15);
}

__device__ __forceinline__ v8f wmma_bf16(v16bf a, v16bf b, v8f c) {
    return __builtin_amdgcn_wmma_f32_16x16x32_bf16(
        /*neg_a=*/false, a, /*neg_b=*/false, b,
        /*c_mod=*/(short)0, c, /*reuse_a=*/false, /*reuse_b=*/false);
}

// ---------------------------------------------------------------------------
// Prep kernels
// ---------------------------------------------------------------------------
__global__ void __launch_bounds__(256) k_convert_x(const float* __restrict__ x,
                                                   bf16_t* __restrict__ xb, int n) {
    int i = blockIdx.x * blockDim.x + threadIdx.x;
    if (i < n) xb[i] = f2bf(x[i]);
}

// Fold W_down into per-head projections:
//   W*_t[(h*64+i)][e] = sum_k W_down[e][h*64+k] * proj[h][i][k]   (*scale for q,k)
__global__ void __launch_bounds__(256) k_prep_qkv_w(const float* __restrict__ Wd,
                             const float* __restrict__ tq,
                             const float* __restrict__ tk,
                             const float* __restrict__ tv,
                             bf16_t* __restrict__ Wq, bf16_t* __restrict__ Wk,
                             bf16_t* __restrict__ Wv) {
    int id = blockIdx.x * blockDim.x + threadIdx.x;      // 3*16*64*1024
    if (id >= 3 * HEADS * SDIM * EMB) return;
    int e = id & (EMB - 1);
    int rest = id >> 10;
    int i = rest & (SDIM - 1);  rest >>= 6;
    int h = rest & (HEADS - 1);
    int which = rest >> 4;                                // 0=q 1=k 2=v
    const float* P = (which == 0) ? tq : (which == 1) ? tk : tv;
    const float* wrow = Wd + (size_t)e * EMB + h * SDIM;  // contiguous 64
    const float* prow = P + ((size_t)h * SDIM + i) * SDIM;
    float acc = 0.f;
#pragma unroll 8
    for (int k = 0; k < SDIM; ++k) acc += wrow[k] * prow[k];
    const float scale = 0.35355339059327373f;             // 64^(-1/4)
    if (which < 2) acc *= scale;
    bf16_t* dst = (which == 0) ? Wq : (which == 1) ? Wk : Wv;
    dst[(size_t)(h * SDIM + i) * EMB + e] = f2bf(acc);
}

__global__ void __launch_bounds__(256) k_prep_unify_w(const float* __restrict__ Wu,
                                                      bf16_t* __restrict__ Wut) {
    int id = blockIdx.x * blockDim.x + threadIdx.x;       // 1024*1024
    if (id >= EMB * EMB) return;
    int n = id & (EMB - 1);
    int e = id >> 10;
    Wut[(size_t)n * EMB + e] = f2bf(Wu[(size_t)e * EMB + n]);
}

// ---------------------------------------------------------------------------
// Stage 1: X[4096,1024] x folded-QKV weights -> Q,K [b][h][t][64], V^T [b][h][64][t]
// One wave computes a 32x64 tile: 8 accumulators (64 VGPRs), 8 WMMAs per 6
// fragment loads, double-buffered over K (step 32, unrolled x2). ~190 VGPRs.
// ---------------------------------------------------------------------------
__global__ void __launch_bounds__(256) k_gemm_qkv(
                           const bf16_t* __restrict__ X,
                           const bf16_t* __restrict__ Wq, const bf16_t* __restrict__ Wk,
                           const bf16_t* __restrict__ Wv,
                           bf16_t* __restrict__ Q, bf16_t* __restrict__ K,
                           bf16_t* __restrict__ Vt) {
    int wave = (blockIdx.x * blockDim.x + threadIdx.x) >> 5;
    int lane = threadIdx.x & 31;
    const int NT = 48;                      // 3072/64 n-tiles
    int n_tile = wave % NT;
    int m_tile = wave / NT;                 // 0..127
    int m0  = m_tile * 32;
    int ng0 = n_tile * 64;                  // global n in [0,3072)
    int which = ng0 >> 10;                  // 0=q 1=k 2=v
    int n0  = ng0 & (EMB - 1);
    const bf16_t* Wt = (which == 0) ? Wq : (which == 1) ? Wk : Wv;

    v8f acc[2][4] = {};
    v16bf a0[2], b0[4], a1[2], b1[4];
#pragma unroll
    for (int r = 0; r < 2; ++r) a0[r] = load_frag_a(X,  m0 + 16 * r, EMB, 0, lane);
#pragma unroll
    for (int c = 0; c < 4; ++c) b0[c] = load_frag_b(Wt, n0 + 16 * c, EMB, 0, lane);

    for (int k0 = 0; k0 < EMB; k0 += 64) {
        // phase-1 loads (k0+32) issued before phase-0 WMMAs
#pragma unroll
        for (int r = 0; r < 2; ++r) a1[r] = load_frag_a(X,  m0 + 16 * r, EMB, k0 + 32, lane);
#pragma unroll
        for (int c = 0; c < 4; ++c) b1[c] = load_frag_b(Wt, n0 + 16 * c, EMB, k0 + 32, lane);
        if (k0 + 128 < EMB) {               // pull panels 2 steps ahead toward WGP
#pragma unroll
            for (int r = 0; r < 2; ++r)
                __builtin_prefetch(X + (size_t)(m0 + 16 * r + (lane & 15)) * EMB + k0 + 128, 0, 3);
#pragma unroll
            for (int c = 0; c < 4; ++c)
                __builtin_prefetch(Wt + (size_t)(n0 + 16 * c + (lane & 15)) * EMB + k0 + 128, 0, 3);
        }
#pragma unroll
        for (int r = 0; r < 2; ++r)
#pragma unroll
            for (int c = 0; c < 4; ++c) acc[r][c] = wmma_bf16(a0[r], b0[c], acc[r][c]);

        int kn = (k0 + 64 < EMB) ? k0 + 64 : 0;   // next phase-0 loads (dummy on last iter)
#pragma unroll
        for (int r = 0; r < 2; ++r) a0[r] = load_frag_a(X,  m0 + 16 * r, EMB, kn, lane);
#pragma unroll
        for (int c = 0; c < 4; ++c) b0[c] = load_frag_b(Wt, n0 + 16 * c, EMB, kn, lane);
#pragma unroll
        for (int r = 0; r < 2; ++r)
#pragma unroll
            for (int c = 0; c < 4; ++c) acc[r][c] = wmma_bf16(a1[r], b1[c], acc[r][c]);
    }

    int half = lane >> 4, cl = lane & 15;
#pragma unroll
    for (int r = 0; r < 2; ++r) {
#pragma unroll
        for (int c = 0; c < 4; ++c) {
#pragma unroll
            for (int i = 0; i < 8; ++i) {
                int row = m0 + 16 * r + i + half * 8;   // global token 0..4095
                int n   = n0 + 16 * c + cl;             // 0..1023
                int h   = n >> 6, sc = n & 63;
                int bb  = row >> 11, tt = row & (SEQ - 1);
                bf16_t v = f2bf(acc[r][c][i]);
                size_t bh = (size_t)(bb * HEADS + h);
                if (which == 0)      Q[(bh * SEQ + tt) * SDIM + sc] = v;
                else if (which == 1) K[(bh * SEQ + tt) * SDIM + sc] = v;
                else                 Vt[(bh * SDIM + sc) * SEQ + tt] = v;
            }
        }
    }
}

// ---------------------------------------------------------------------------
// Stage 2: flash attention. One wave handles 32 query rows of one (b,h).
// Scores in WMMA C-regs; online softmax via 16-lane shfl_xor reductions;
// V fragments issued after score WMMAs (K frags dead) and consumed after the
// softmax VALU block; P re-laid out C-frag -> A-frag through LDS.
// ---------------------------------------------------------------------------
__global__ void __launch_bounds__(128) k_attn(
                       const bf16_t* __restrict__ Q, const bf16_t* __restrict__ K,
                       const bf16_t* __restrict__ Vt, bf16_t* __restrict__ att) {
    __shared__ bf16_t Psh[4][32][32];           // per-wave P staging (2 KB each)
    int wv   = threadIdx.x >> 5;                // wave in block: 0..3
    int lane = threadIdx.x & 31;
    int gw   = blockIdx.x * 4 + wv;             // 0..2047
    int qt   = gw & 63;                         // 64 q-tiles of 32 rows
    int bh   = gw >> 6;                         // b*16+h
    int q0   = qt * 32;
    const bf16_t* Qp = Q  + (size_t)bh * SEQ * SDIM;
    const bf16_t* Kp = K  + (size_t)bh * SEQ * SDIM;
    const bf16_t* Vp = Vt + (size_t)bh * SDIM * SEQ;
    int half = lane >> 4, cl = lane & 15;

    // Q fragments: [row-block][feature-half], reused across all KV blocks
    v16bf qa[2][2];
#pragma unroll
    for (int rb = 0; rb < 2; ++rb) {
        qa[rb][0] = load_frag_a(Qp, q0 + rb * 16, SDIM, 0,  lane);
        qa[rb][1] = load_frag_a(Qp, q0 + rb * 16, SDIM, 32, lane);
    }

    v8f o[2][4] = {};
    float m[2][8], l[2][8];
#pragma unroll
    for (int rb = 0; rb < 2; ++rb)
#pragma unroll
        for (int i = 0; i < 8; ++i) { m[rb][i] = -1e30f; l[rb][i] = 0.f; }

    for (int kv = 0; kv < SEQ; kv += 32) {
        // K^T fragments (2 kv-column blocks x 2 feature halves)
        v16bf kb00 = load_frag_b(Kp, kv,      SDIM, 0,  lane);
        v16bf kb01 = load_frag_b(Kp, kv,      SDIM, 32, lane);
        v16bf kb10 = load_frag_b(Kp, kv + 16, SDIM, 0,  lane);
        v16bf kb11 = load_frag_b(Kp, kv + 16, SDIM, 32, lane);

        // S = Q @ K^T : [row-block][col-block], 8 WMMAs
        v8f s[2][2] = {};
#pragma unroll
        for (int rb = 0; rb < 2; ++rb) {
            s[rb][0] = wmma_bf16(qa[rb][0], kb00, s[rb][0]);
            s[rb][0] = wmma_bf16(qa[rb][1], kb01, s[rb][0]);
            s[rb][1] = wmma_bf16(qa[rb][0], kb10, s[rb][1]);
            s[rb][1] = wmma_bf16(qa[rb][1], kb11, s[rb][1]);
        }
        // V fragments for this kv block: K frags are dead now; loads overlap softmax
        v16bf vb[4];
#pragma unroll
        for (int f = 0; f < 4; ++f) vb[f] = load_frag_b(Vp, f * 16, SEQ, kv, lane);

        // Online softmax. Row r's 16 columns live in lanes 0-15 (or 16-31).
#pragma unroll
        for (int rb = 0; rb < 2; ++rb) {
#pragma unroll
            for (int i = 0; i < 8; ++i) {
                float v = fmaxf(s[rb][0][i], s[rb][1][i]);
#pragma unroll
                for (int mk = 1; mk < 16; mk <<= 1) v = fmaxf(v, __shfl_xor(v, mk, 32));
                float mn    = fmaxf(m[rb][i], v);
                float alpha = __expf(m[rb][i] - mn);
                float p0 = __expf(s[rb][0][i] - mn);
                float p1 = __expf(s[rb][1][i] - mn);
                float rs = p0 + p1;
#pragma unroll
                for (int mk = 1; mk < 16; mk <<= 1) rs += __shfl_xor(rs, mk, 32);
                l[rb][i] = l[rb][i] * alpha + rs;
                m[rb][i] = mn;
#pragma unroll
                for (int f = 0; f < 4; ++f) o[rb][f][i] *= alpha;
                Psh[wv][rb * 16 + i + half * 8][cl]      = f2bf(p0);
                Psh[wv][rb * 16 + i + half * 8][cl + 16] = f2bf(p1);
            }
        }
        asm volatile("s_wait_dscnt 0x0" ::: "memory");
        // Reload P as A-fragments (rows = q, K = 32 kv tokens)
        v16bf pa[2];
#pragma unroll
        for (int rb = 0; rb < 2; ++rb) {
            const bf16_t* pp = &Psh[wv][rb * 16 + cl][half * 8];
            v8bf lo = *(const v8bf*)(pp);
            v8bf hi = *(const v8bf*)(pp + 16);
            pa[rb] = __builtin_shufflevector(lo, hi, 0,1,2,3,4,5,6,7,8,9,10,11,12,13,14,15);
        }
        asm volatile("" ::: "memory");   // keep LDS reads before next iter's stores
        // O += P @ V
#pragma unroll
        for (int rb = 0; rb < 2; ++rb)
#pragma unroll
            for (int f = 0; f < 4; ++f) o[rb][f] = wmma_bf16(pa[rb], vb[f], o[rb][f]);
    }
    // Epilogue: normalize rows and write bf16 att_out[token][h*64+feature]
    int h = bh & (HEADS - 1), bb = bh >> 4;
#pragma unroll
    for (int rb = 0; rb < 2; ++rb) {
#pragma unroll
        for (int f = 0; f < 4; ++f) {
#pragma unroll
            for (int i = 0; i < 8; ++i) {
                int row  = q0 + rb * 16 + i + half * 8;
                int gtok = bb * SEQ + row;
                float v  = o[rb][f][i] / l[rb][i];
                att[(size_t)gtok * EMB + h * SDIM + f * 16 + cl] = f2bf(v);
            }
        }
    }
}

// ---------------------------------------------------------------------------
// Stage 3: out = att @ W_unify + b_unify (f32 output), 32x64 tile/wave,
// double-buffered K loop.
// ---------------------------------------------------------------------------
__global__ void __launch_bounds__(256) k_gemm_unify(
                             const bf16_t* __restrict__ A, const bf16_t* __restrict__ WuT,
                             const float* __restrict__ bias, float* __restrict__ out) {
    int wave = (blockIdx.x * blockDim.x + threadIdx.x) >> 5;
    int lane = threadIdx.x & 31;
    int n_tile = wave & 15;                 // 1024/64
    int m_tile = wave >> 4;                 // 0..127
    int m0 = m_tile * 32, n0 = n_tile * 64;

    v8f acc[2][4] = {};
    v16bf a0[2], b0[4], a1[2], b1[4];
#pragma unroll
    for (int r = 0; r < 2; ++r) a0[r] = load_frag_a(A,   m0 + 16 * r, EMB, 0, lane);
#pragma unroll
    for (int c = 0; c < 4; ++c) b0[c] = load_frag_b(WuT, n0 + 16 * c, EMB, 0, lane);

    for (int k0 = 0; k0 < EMB; k0 += 64) {
#pragma unroll
        for (int r = 0; r < 2; ++r) a1[r] = load_frag_a(A,   m0 + 16 * r, EMB, k0 + 32, lane);
#pragma unroll
        for (int c = 0; c < 4; ++c) b1[c] = load_frag_b(WuT, n0 + 16 * c, EMB, k0 + 32, lane);
        if (k0 + 128 < EMB) {
#pragma unroll
            for (int r = 0; r < 2; ++r)
                __builtin_prefetch(A + (size_t)(m0 + 16 * r + (lane & 15)) * EMB + k0 + 128, 0, 3);
#pragma unroll
            for (int c = 0; c < 4; ++c)
                __builtin_prefetch(WuT + (size_t)(n0 + 16 * c + (lane & 15)) * EMB + k0 + 128, 0, 3);
        }
#pragma unroll
        for (int r = 0; r < 2; ++r)
#pragma unroll
            for (int c = 0; c < 4; ++c) acc[r][c] = wmma_bf16(a0[r], b0[c], acc[r][c]);

        int kn = (k0 + 64 < EMB) ? k0 + 64 : 0;
#pragma unroll
        for (int r = 0; r < 2; ++r) a0[r] = load_frag_a(A,   m0 + 16 * r, EMB, kn, lane);
#pragma unroll
        for (int c = 0; c < 4; ++c) b0[c] = load_frag_b(WuT, n0 + 16 * c, EMB, kn, lane);
#pragma unroll
        for (int r = 0; r < 2; ++r)
#pragma unroll
            for (int c = 0; c < 4; ++c) acc[r][c] = wmma_bf16(a1[r], b1[c], acc[r][c]);
    }

    int half = lane >> 4, cl = lane & 15;
#pragma unroll
    for (int r = 0; r < 2; ++r) {
#pragma unroll
        for (int c = 0; c < 4; ++c) {
#pragma unroll
            for (int i = 0; i < 8; ++i) {
                int row = m0 + 16 * r + i + half * 8;
                int n   = n0 + 16 * c + cl;
                out[(size_t)row * EMB + n] = acc[r][c][i] + bias[n];
            }
        }
    }
}

// ---------------------------------------------------------------------------
extern "C" void kernel_launch(void* const* d_in, const int* in_sizes, int n_in,
                              void* d_out, int out_size, void* d_ws, size_t ws_size,
                              hipStream_t stream) {
    const float* x  = (const float*)d_in[0];   // [2,2048,1024]
    const float* Wd = (const float*)d_in[1];   // [1024,1024]
    const float* tk = (const float*)d_in[2];   // tokeys    [16,64,64]
    const float* tq = (const float*)d_in[3];   // toqueries [16,64,64]
    const float* tv = (const float*)d_in[4];   // tovalues  [16,64,64]
    const float* Wu = (const float*)d_in[5];   // [1024,1024]
    const float* bu = (const float*)d_in[6];   // [1024]
    float* out = (float*)d_out;

    // Workspace carve (all 256B-aligned sizes), total = 48 MB
    char* w = (char*)d_ws;
    bf16_t* x_bf = (bf16_t*)w;  w += (size_t)NTOK * EMB * 2;            // 8 MB
    bf16_t* Wq   = (bf16_t*)w;  w += (size_t)EMB * EMB * 2;             // 2 MB
    bf16_t* Wk   = (bf16_t*)w;  w += (size_t)EMB * EMB * 2;             // 2 MB
    bf16_t* Wv   = (bf16_t*)w;  w += (size_t)EMB * EMB * 2;             // 2 MB
    bf16_t* WuT  = (bf16_t*)w;  w += (size_t)EMB * EMB * 2;             // 2 MB
    bf16_t* Qb   = (bf16_t*)w;  w += (size_t)NTOK * EMB * 2;            // 8 MB
    bf16_t* Kb   = (bf16_t*)w;  w += (size_t)NTOK * EMB * 2;            // 8 MB
    bf16_t* Vtb  = (bf16_t*)w;  w += (size_t)NTOK * EMB * 2;            // 8 MB
    bf16_t* att  = (bf16_t*)w;  w += (size_t)NTOK * EMB * 2;            // 8 MB

    // Prep
    k_convert_x<<<(NTOK * EMB) / 256, 256, 0, stream>>>(x, x_bf, NTOK * EMB);
    k_prep_qkv_w<<<(3 * HEADS * SDIM * EMB) / 256, 256, 0, stream>>>(
        Wd, tq, tk, tv, Wq, Wk, Wv);
    k_prep_unify_w<<<(EMB * EMB) / 256, 256, 0, stream>>>(Wu, WuT);

    // Stage 1: QKV projection (128 m-tiles x 48 n-tiles = 6144 waves)
    k_gemm_qkv<<<6144 / 8, 256, 0, stream>>>(x_bf, Wq, Wk, Wv, Qb, Kb, Vtb);

    // Stage 2: flash attention (2048 waves, 4 per block)
    k_attn<<<512, 128, 0, stream>>>(Qb, Kb, Vtb, att);

    // Stage 3: unify (128 x 16 = 2048 waves)
    k_gemm_unify<<<2048 / 8, 256, 0, stream>>>(att, WuT, bu, out);
}
